// GraphAttentionNet_6717328851512
// MI455X (gfx1250) — compile-verified, matrix operand order checked by reading, not run on hardware
//
#include <hip/hip_runtime.h>
#include <hip/hip_bf16.h>

typedef __attribute__((ext_vector_type(16))) _Float16 v16h;
typedef __attribute__((ext_vector_type(8)))  _Float16 v8h;
typedef __attribute__((ext_vector_type(8)))  float    v8f;

#define B_    8
#define T_    64
#define N_    256
#define FIN_  128
#define FOUT_ 64
#define NEG_INF_F (-9.0e15f)

#define WT_STRIDE  136   // halves: FIN + 8 pad   (W^T staging)
#define WHT_STRIDE 264   // halves: N   + 8 pad   (Wh^T)
#define ACH_STRIDE 40    // halves: 32  + 8 pad   (alpha chunk)

__device__ __forceinline__ v8h ldg_f32x8_to_h(const float* p) {
  float4 x = ((const float4*)p)[0];
  float4 y = ((const float4*)p)[1];
  v8h r;
  r[0]=(_Float16)x.x; r[1]=(_Float16)x.y; r[2]=(_Float16)x.z; r[3]=(_Float16)x.w;
  r[4]=(_Float16)y.x; r[5]=(_Float16)y.y; r[6]=(_Float16)y.z; r[7]=(_Float16)y.w;
  return r;
}

__device__ __forceinline__ v16h cat16(v8h lo, v8h hi) {
  return __builtin_shufflevector(lo, hi, 0,1,2,3,4,5,6,7,8,9,10,11,12,13,14,15);
}

__global__ __launch_bounds__(256) void gat_kernel(
    const float* __restrict__ h, const float* __restrict__ W,
    const float* __restrict__ a, const float* __restrict__ adj,
    float* __restrict__ out)
{
  // Phase A: W^T f16 [64][136]  /  Phase C: alpha chunks 8 waves * 16 * 40 (5120h)
  __shared__ __align__(16) _Float16 sWTaC[FOUT_ * WT_STRIDE];      // 17408 B
  __shared__ __align__(16) _Float16 sWhT[FOUT_ * WHT_STRIDE];      // 33792 B  Wh^T f16 [n][m]
  __shared__ float sWh1[N_], sWh2[N_];                             //  2048 B
  __shared__ float sRowMax[128], sRowInv[128];                     //  1024 B (8 waves * 16 rows)

  const int tid  = threadIdx.x;
  const int lane = tid & 31;
  const int wave = tid >> 5;
  const int bt   = blockIdx.x;                 // 0..511 == (b,t)
  const size_t hBase = (size_t)bt * N_ * FIN_;

  const int rsel = lane & 15;                  // row/col within 16-wide tile
  const int ksel = (lane >> 4) * 8;            // half-wave K offset for 16-bit A/B frags

  // ---------------- Phase A0: stage W^T (f16) ----------------
  for (int idx = tid; idx < FIN_ * FOUT_; idx += 256) {
    int k = idx >> 6, n = idx & 63;            // W is [FIN][FOUT] row-major
    sWTaC[n * WT_STRIDE + k] = (_Float16)W[idx];
  }
  __syncthreads();

  // ---------------- Phase A1: GEMM1  Wh = h * W  -> Wh^T (LDS, f16) ----------------
  for (int rt = wave * 2; rt < wave * 2 + 2; ++rt) {
    const int m0 = rt * 16;
    v8f cacc[4] = {};
    #pragma unroll
    for (int kk = 0; kk < 4; ++kk) {
      const float* ap = h + hBase + (size_t)(m0 + rsel) * FIN_ + kk * 32 + ksel;
      v16h afrag = cat16(ldg_f32x8_to_h(ap), ldg_f32x8_to_h(ap + 16));
      const int kb = kk * 32 + ksel;
      #pragma unroll
      for (int ct = 0; ct < 4; ++ct) {
        const _Float16* bp = &sWTaC[(ct * 16 + rsel) * WT_STRIDE + kb];
        v16h bfrag = cat16(*(const v8h*)bp, *(const v8h*)(bp + 16));
        cacc[ct] = __builtin_amdgcn_wmma_f32_16x16x32_f16(
            false, afrag, false, bfrag, (short)0, cacc[ct], false, false);
      }
    }
    #pragma unroll
    for (int ct = 0; ct < 4; ++ct) {
      #pragma unroll
      for (int v = 0; v < 8; ++v) {
        int node = m0 + v + 8 * (lane >> 4);   // C layout: VGPR v, lane -> (row, col)
        int feat = ct * 16 + rsel;
        sWhT[feat * WHT_STRIDE + node] = (_Float16)cacc[ct][v];
      }
    }
  }
  __syncthreads();

  // ---------------- Phase B: Wh1 = Wh*a1, Wh2 = Wh*a2 (f32) ----------------
  {
    const int m = tid;
    float s1 = 0.f, s2 = 0.f;
    #pragma unroll 4
    for (int n = 0; n < FOUT_; ++n) {
      float whv = (float)sWhT[n * WHT_STRIDE + m];
      s1 += whv * a[n];
      s2 += whv * a[FOUT_ + n];
    }
    sWh1[m] = s1; sWh2[m] = s2;
  }
  __syncthreads();

  // ---------------- Phase C: masked softmax + GEMM2 (alpha * Wh) + ELU ----------------
  _Float16* aCh = sWTaC + wave * (16 * ACH_STRIDE);   // W^T region reused per-wave
  for (int rt = wave * 2; rt < wave * 2 + 2; ++rt) {
    const int i0 = rt * 16;

    // pass 1: per-row online softmax stats (max, sum)
    for (int r = 0; r < 16; ++r) {
      const int i = i0 + r;
      const float wh1 = sWh1[i];
      const float* adjRow = adj + (size_t)i * N_;
      float mrun = -3.0e38f, srun = 0.f;
      #pragma unroll
      for (int q = 0; q < 8; ++q) {
        int j = lane + 32 * q;
        float ev = wh1 + sWh2[j];
        ev = ev > 0.f ? ev : 0.01f * ev;                 // leaky_relu(0.01)
        float mk = (adjRow[j] > 0.f) ? ev : NEG_INF_F;   // mask
        if (mk > mrun) { srun = srun * __expf(mrun - mk) + 1.f; mrun = mk; }
        else           { srun += __expf(mk - mrun); }
      }
      #pragma unroll
      for (int off = 16; off > 0; off >>= 1) {
        float mo = __shfl_xor(mrun, off);
        float so = __shfl_xor(srun, off);
        float mn = fmaxf(mrun, mo);
        srun = srun * __expf(mrun - mn) + so * __expf(mo - mn);
        mrun = mn;
      }
      if (lane == 0) { sRowMax[wave * 16 + r] = mrun; sRowInv[wave * 16 + r] = 1.f / srun; }
    }

    // pass 2: chunked alpha recompute (16x32 f16) fused into WMMA GEMM2
    v8f cacc[4] = {};
    for (int kk = 0; kk < 8; ++kk) {
      const int j0 = kk * 32;
      const int j  = j0 + lane;
      const float wh2j = sWh2[j];
      #pragma unroll
      for (int r = 0; r < 16; ++r) {
        const int i = i0 + r;
        float ev = sWh1[i] + wh2j;
        ev = ev > 0.f ? ev : 0.01f * ev;
        float alpha = (adj[(size_t)i * N_ + j] > 0.f)
            ? __expf(ev - sRowMax[wave * 16 + r]) * sRowInv[wave * 16 + r]
            : 0.f;
        aCh[r * ACH_STRIDE + lane] = (_Float16)alpha;
      }
      // A frag: 16 rows of alpha chunk
      const _Float16* apc = &aCh[rsel * ACH_STRIDE + ksel];
      v16h afrag = cat16(*(const v8h*)apc, *(const v8h*)(apc + 16));
      // B frags: Wh^T, K runs over nodes m
      const int kb = j0 + ksel;
      #pragma unroll
      for (int ct = 0; ct < 4; ++ct) {
        const _Float16* bp = &sWhT[(ct * 16 + rsel) * WHT_STRIDE + kb];
        v16h bfrag = cat16(*(const v8h*)bp, *(const v8h*)(bp + 16));
        cacc[ct] = __builtin_amdgcn_wmma_f32_16x16x32_f16(
            false, afrag, false, bfrag, (short)0, cacc[ct], false, false);
      }
    }

    // epilogue: ELU, coalesced f32 stores
    #pragma unroll
    for (int ct = 0; ct < 4; ++ct) {
      #pragma unroll
      for (int v = 0; v < 8; ++v) {
        int i = i0 + v + 8 * (lane >> 4);
        int n = ct * 16 + rsel;
        float x = cacc[ct][v];
        float y = x > 0.f ? x : (__expf(x) - 1.f);       // elu(alpha=1)
        out[((size_t)bt * N_ + i) * FOUT_ + n] = y;
      }
    }
  }
}

extern "C" void kernel_launch(void* const* d_in, const int* in_sizes, int n_in,
                              void* d_out, int out_size, void* d_ws, size_t ws_size,
                              hipStream_t stream) {
  (void)in_sizes; (void)n_in; (void)out_size; (void)d_ws; (void)ws_size;
  const float* h   = (const float*)d_in[0];
  const float* W   = (const float*)d_in[1];
  const float* a   = (const float*)d_in[2];
  const float* adj = (const float*)d_in[3];
  float* out = (float*)d_out;
  hipLaunchKernelGGL(gat_kernel, dim3(B_ * T_), dim3(256), 0, stream,
                     h, W, a, adj, out);
}
